// BigReimplementationPathIntegrator_1400159338671
// MI455X (gfx1250) — matrix operand order repaired and checked by reading, not compiled, not run on hardware
//
#include <hip/hip_runtime.h>
#include <hip/hip_bf16.h>
#include <stdint.h>

// ---------------------------------------------------------------------------
// MI455X (gfx1250) implementation.
//  - convs: direct VALU (small fraction of FLOPs)
//  - fc1/fc2/fc3: bf16 WMMA GEMM (v_wmma_f32_16x16x32_bf16), f32 accum
//  - GRU: single persistent kernel, 16 WGs, w_hh staged once in LDS (bf16),
//    grid barrier per step via device atomics; gates computed in-register
//    straight from WMMA accumulators.
// ---------------------------------------------------------------------------

typedef __attribute__((ext_vector_type(16))) __bf16 v16bf;
typedef __attribute__((ext_vector_type(8)))  float  v8f;

union BF16Frag {
  v16bf v;
  unsigned short u[16];
  uint4 q[2];
};

__device__ __forceinline__ unsigned short f2bf_bits(float f) {
  unsigned int u = __builtin_bit_cast(unsigned int, f);
  unsigned int r = (u + 0x7FFFu + ((u >> 16) & 1u)) >> 16;  // RNE
  return (unsigned short)r;
}

__device__ __forceinline__ void pack8(unsigned short* d, float4 a, float4 b) {
  d[0] = f2bf_bits(a.x); d[1] = f2bf_bits(a.y);
  d[2] = f2bf_bits(a.z); d[3] = f2bf_bits(a.w);
  d[4] = f2bf_bits(b.x); d[5] = f2bf_bits(b.y);
  d[6] = f2bf_bits(b.z); d[7] = f2bf_bits(b.w);
}

// ---------------------------------------------------------------------------
// Direct conv + ReLU (stride 3, pad 2). One thread per output element.
// ---------------------------------------------------------------------------
__global__ void conv_relu_kernel(const float* __restrict__ in,
                                 const float* __restrict__ w,
                                 const float* __restrict__ bias,
                                 float* __restrict__ out,
                                 int NB, int IC, int IH, int IW,
                                 int OC, int OH, int OW, int KS) {
  int idx = blockIdx.x * blockDim.x + threadIdx.x;
  int total = NB * OC * OH * OW;
  if (idx >= total) return;
  int ox = idx % OW;
  int oy = (idx / OW) % OH;
  int oc = (idx / (OW * OH)) % OC;
  int b  = idx / (OW * OH * OC);
  float acc = bias[oc];
  int iy0 = oy * 3 - 2;
  int ix0 = ox * 3 - 2;
  const float* wp = w + (size_t)oc * IC * KS * KS;
  const float* ip = in + (size_t)b * IC * IH * IW;
  for (int ic = 0; ic < IC; ++ic) {
    const float* ipc = ip + (size_t)ic * IH * IW;
    const float* wpc = wp + ic * KS * KS;
    for (int ky = 0; ky < KS; ++ky) {
      int iy = iy0 + ky;
      if ((unsigned)iy >= (unsigned)IH) continue;
      for (int kx = 0; kx < KS; ++kx) {
        int ix = ix0 + kx;
        if ((unsigned)ix >= (unsigned)IW) continue;
        acc = fmaf(ipc[iy * IW + ix], wpc[ky * KS + kx], acc);
      }
    }
  }
  out[idx] = fmaxf(acc, 0.0f);
}

// ---------------------------------------------------------------------------
// FC GEMM: Y[M,N] = act(X[M,K] @ W[N,K]^T + bias). One wave per 16x16 tile.
// bf16 WMMA, K multiple of 32, M/N multiple of 16. blockDim = 32.
// ---------------------------------------------------------------------------
__global__ void fc_wmma_kernel(const float* __restrict__ X,
                               const float* __restrict__ W,
                               const float* __restrict__ bias,
                               float* __restrict__ Y,
                               int K, int N, int do_relu) {
  const int lane = threadIdx.x & 31;
  const int n0 = blockIdx.x * 16;
  const int m0 = blockIdx.y * 16;
  const int nl = lane & 15;
  const int kh = (lane >> 4) * 8;    // A k-half offset
  const int kb = (lane >> 4) * 16;   // unused here; B also uses kh
  const float* xrow = X + (size_t)(m0 + nl) * K;
  const float* wrow = W + (size_t)(n0 + nl) * K;

  v8f acc = {};
  for (int k0 = 0; k0 < K; k0 += 32) {
    BF16Frag a, b;
    // A 16x32 fragment: lane row = m0+nl, k = k0+kh+{0..7} and k0+16+kh+{0..7}
    float4 x0 = *(const float4*)(xrow + k0 + kh);
    float4 x1 = *(const float4*)(xrow + k0 + kh + 4);
    float4 x2 = *(const float4*)(xrow + k0 + 16 + kh);
    float4 x3 = *(const float4*)(xrow + k0 + 16 + kh + 4);
    pack8(a.u, x0, x1);
    pack8(a.u + 8, x2, x3);
    // B 32x16 fragment: lane col = n0+nl, k = k0+kh+{0..7} and k0+16+kh+{0..7}
    float4 w0 = *(const float4*)(wrow + k0 + kh);
    float4 w1 = *(const float4*)(wrow + k0 + kh + 4);
    float4 w2 = *(const float4*)(wrow + k0 + 16 + kh);
    float4 w3 = *(const float4*)(wrow + k0 + 16 + kh + 4);
    pack8(b.u, w0, w1);
    pack8(b.u + 8, w2, w3);
    acc = __builtin_amdgcn_wmma_f32_16x16x32_bf16(
        false, a.v, false, b.v, (short)0, acc, false, false);
  }

  const int ncol = n0 + nl;
  const float bz = bias[ncol];
  const int rbase = m0 + 8 * (lane >> 4);
#pragma unroll
  for (int r = 0; r < 8; ++r) {
    float v = acc[r] + bz;
    if (do_relu) v = fmaxf(v, 0.0f);
    Y[(size_t)(rbase + r) * N + ncol] = v;
  }
}

// ---------------------------------------------------------------------------
// Persistent GRU. Grid = 16 WGs (32 hidden cols each), 512 threads (16 waves).
// Wave (wid) owns M-tile rows 16*(wid&7).. and gate-column half p = wid>>3.
// Each step: gh = h @ w_hh^T via WMMA (w_hh slice in LDS as bf16),
// gates in registers, write hy to ping-pong h buffer + output, grid barrier.
// ---------------------------------------------------------------------------
#define GB   128
#define GH   512
#define GT   100
#define GNWG 16
#define GHB  32
#define WHH_STRIDE 520   // 512 + 8 bf16 pad (bank spread)

__global__ void gru_persistent_kernel(float* __restrict__ hA,
                                      float* __restrict__ hB,
                                      const float* __restrict__ actions, // [B,T,2]
                                      const float* __restrict__ w_ih,    // [1536,2]
                                      const float* __restrict__ w_hh,    // [1536,512]
                                      const float* __restrict__ b_ih,
                                      const float* __restrict__ b_hh,
                                      float* __restrict__ out,           // [B,T,512]
                                      float* __restrict__ hT,            // [B,512]
                                      unsigned* __restrict__ cnt) {
  extern __shared__ unsigned short lds_whh[];  // [96][WHH_STRIDE] bf16 bits
  const int tid  = threadIdx.x;
  const int lane = tid & 31;
  const int wid  = tid >> 5;            // 0..15
  const int hb   = blockIdx.x * GHB;    // base hidden column of this WG

  // Stage this WG's 96 w_hh rows (r,z,n x 32 cols) as bf16 into LDS, once.
  for (int i = tid; i < 96 * GH; i += blockDim.x) {
    int q = i / GH;                     // local gate col 0..95
    int k = i % GH;
    int gate = q >> 5;
    int jj = q & 31;
    int row = gate * GH + hb + jj;      // row of w_hh [1536,512]
    lds_whh[q * WHH_STRIDE + k] = f2bf_bits(w_hh[(size_t)row * GH + k]);
  }
  __syncthreads();

  const int m0 = 16 * (wid & 7);        // M-tile base row
  const int p  = wid >> 3;              // which 16-col half of the 32 cols
  const int nl = lane & 15;
  const int kh = (lane >> 4) * 8;
  const int kb = (lane >> 4) * 16;
  const int arow = m0 + nl;             // A-fragment row for this lane
  const int col  = hb + p * 16 + nl;    // hidden column handled in epilogue

  // Step-invariant gate parameters for this lane's column.
  const float wi_r0 = w_ih[2 * col + 0],          wi_r1 = w_ih[2 * col + 1];
  const float wi_z0 = w_ih[2 * (GH + col) + 0],   wi_z1 = w_ih[2 * (GH + col) + 1];
  const float wi_n0 = w_ih[2 * (2*GH + col) + 0], wi_n1 = w_ih[2 * (2*GH + col) + 1];
  const float bi_r = b_ih[col], bi_z = b_ih[GH + col], bi_n = b_ih[2*GH + col];
  const float bh_r = b_hh[col], bh_z = b_hh[GH + col], bh_n = b_hh[2*GH + col];

  // LDS row bases for the three gate fragments (q = gate*32 + p*16 + nl).
  const unsigned short* ldr = &lds_whh[(size_t)(p * 16 + nl) * WHH_STRIDE];
  const unsigned short* ldz = &lds_whh[(size_t)(32 + p * 16 + nl) * WHH_STRIDE];
  const unsigned short* ldn = &lds_whh[(size_t)(64 + p * 16 + nl) * WHH_STRIDE];

  for (int t = 0; t < GT; ++t) {
    const float* cur = (t & 1) ? hB : hA;
    float*       nxt = (t & 1) ? hA : hB;

    v8f accr = {}, accz = {}, accn = {};
    const float* hrow = cur + (size_t)arow * GH;
    for (int k0 = 0; k0 < GH; k0 += 32) {
      if (k0 + 32 < GH) __builtin_prefetch(hrow + k0 + 32 + kh, 0, 1);
      BF16Frag a;
      float4 x0 = *(const float4*)(hrow + k0 + kb);
      float4 x1 = *(const float4*)(hrow + k0 + kb + 4);
      float4 x2 = *(const float4*)(hrow + k0 + kb + 8);
      float4 x3 = *(const float4*)(hrow + k0 + kb + 12);
      pack8(a.u, x0, x1);
      pack8(a.u + 8, x2, x3);
      BF16Frag br, bz, bn;
      const uint4* pr = (const uint4*)(ldr + k0 + kb);
      const uint4* pz = (const uint4*)(ldz + k0 + kb);
      const uint4* pn = (const uint4*)(ldn + k0 + kb);
      br.q[0] = pr[0]; br.q[1] = pr[1];
      bz.q[0] = pz[0]; bz.q[1] = pz[1];
      bn.q[0] = pn[0]; bn.q[1] = pn[1];
      accr = __builtin_amdgcn_wmma_f32_16x16x32_bf16(
          false, a.v, false, br.v, (short)0, accr, false, false);
      accz = __builtin_amdgcn_wmma_f32_16x16x32_bf16(
          false, a.v, false, bz.v, (short)0, accz, false, false);
      accn = __builtin_amdgcn_wmma_f32_16x16x32_bf16(
          false, a.v, false, bn.v, (short)0, accn, false, false);
    }

    // Gate epilogue: r/z/n for (m,col) live in the same lane + VGPR slot.
    const int rbase = m0 + 8 * (lane >> 4);
#pragma unroll
    for (int r = 0; r < 8; ++r) {
      int m = rbase + r;
      float a0 = actions[((size_t)m * GT + t) * 2 + 0];
      float a1 = actions[((size_t)m * GT + t) * 2 + 1];
      float gi_r = fmaf(a0, wi_r0, fmaf(a1, wi_r1, bi_r));
      float gi_z = fmaf(a0, wi_z0, fmaf(a1, wi_z1, bi_z));
      float gi_n = fmaf(a0, wi_n0, fmaf(a1, wi_n1, bi_n));
      float gh_r = accr[r] + bh_r;
      float gh_z = accz[r] + bh_z;
      float gh_n = accn[r] + bh_n;
      float rg = 1.0f / (1.0f + __expf(-(gi_r + gh_r)));
      float zg = 1.0f / (1.0f + __expf(-(gi_z + gh_z)));
      float ng = tanhf(gi_n + rg * gh_n);
      float hv = cur[(size_t)m * GH + col];
      float hy = ng + zg * (hv - ng);
      nxt[(size_t)m * GH + col] = hy;
      out[(size_t)m * (GT * GH) + (size_t)t * GH + col] = hy;
      if (t == GT - 1) hT[(size_t)m * GH + col] = hy;
    }

    // Grid-wide barrier (16 co-resident WGs; counter monotonically increases).
    __syncthreads();
    if (tid == 0) {
      __threadfence();
      atomicAdd(cnt, 1u);
      unsigned target = (unsigned)(GNWG * (t + 1));
      while (*((volatile unsigned*)cnt) < target) {
        __builtin_amdgcn_s_sleep(2);
      }
      __threadfence();
    }
    __syncthreads();
  }
}

__global__ void init_kernel(unsigned* cnt) {
  if (threadIdx.x == 0) *cnt = 0u;
}

// ---------------------------------------------------------------------------
extern "C" void kernel_launch(void* const* d_in, const int* in_sizes, int n_in,
                              void* d_out, int out_size, void* d_ws, size_t ws_size,
                              hipStream_t stream) {
  const float* images  = (const float*)d_in[0];
  const float* actions = (const float*)d_in[1];
  const float* cw1 = (const float*)d_in[2];
  const float* cb1 = (const float*)d_in[3];
  const float* cw2 = (const float*)d_in[4];
  const float* cb2 = (const float*)d_in[5];
  const float* cw3 = (const float*)d_in[6];
  const float* cb3 = (const float*)d_in[7];
  const float* fw1 = (const float*)d_in[8];
  const float* fb1 = (const float*)d_in[9];
  const float* fw2 = (const float*)d_in[10];
  const float* fb2 = (const float*)d_in[11];
  const float* fw3 = (const float*)d_in[12];
  const float* fb3 = (const float*)d_in[13];
  const float* w_ih = (const float*)d_in[14];
  const float* w_hh = (const float*)d_in[15];
  const float* b_ih = (const float*)d_in[16];
  const float* b_hh = (const float*)d_in[17];

  char* ws = (char*)d_ws;
  size_t off = 0;
  auto alloc = [&](size_t bytes) -> void* {
    void* ptr = ws + off;
    off = (off + bytes + 255) & ~(size_t)255;
    return ptr;
  };

  float* c1   = (float*)alloc(128ull * 64 * 24 * 24 * 4);
  float* c2   = (float*)alloc(128ull * 64 * 9 * 9 * 4);
  float* flat = (float*)alloc(128ull * 1024 * 4);
  float* f1   = (float*)alloc(128ull * 1024 * 4);
  float* f2   = (float*)alloc(128ull * 512 * 4);
  float* hA   = (float*)alloc(128ull * 512 * 4);
  float* hB   = (float*)alloc(128ull * 512 * 4);
  unsigned* cnt = (unsigned*)alloc(256);

  float* out = (float*)d_out;                 // [128,100,512]
  float* hT  = out + 128ull * 100 * 512;      // [128,512]

  init_kernel<<<1, 64, 0, stream>>>(cnt);

  {
    int total = 128 * 64 * 24 * 24;
    conv_relu_kernel<<<(total + 255) / 256, 256, 0, stream>>>(
        images, cw1, cb1, c1, 128, 3, 72, 72, 64, 24, 24, 5);
  }
  {
    int total = 128 * 64 * 9 * 9;
    conv_relu_kernel<<<(total + 255) / 256, 256, 0, stream>>>(
        c1, cw2, cb2, c2, 128, 64, 24, 24, 64, 9, 9, 3);
  }
  {
    int total = 128 * 64 * 4 * 4;
    conv_relu_kernel<<<(total + 255) / 256, 256, 0, stream>>>(
        c2, cw3, cb3, flat, 128, 64, 9, 9, 64, 4, 4, 3);
  }

  fc_wmma_kernel<<<dim3(1024 / 16, 128 / 16), 32, 0, stream>>>(
      flat, fw1, fb1, f1, 1024, 1024, 1);
  fc_wmma_kernel<<<dim3(512 / 16, 128 / 16), 32, 0, stream>>>(
      f1, fw2, fb2, f2, 1024, 512, 1);
  fc_wmma_kernel<<<dim3(512 / 16, 128 / 16), 32, 0, stream>>>(
      f2, fw3, fb3, hA, 512, 512, 0);

  size_t lds_bytes = 96ull * WHH_STRIDE * 2;  // ~100 KB, within 320 KB WGP LDS
  gru_persistent_kernel<<<GNWG, 512, lds_bytes, stream>>>(
      hA, hB, actions, w_ih, w_hh, b_ih, b_hh, out, hT, cnt);
}
